// NAM_33655363732260
// MI455X (gfx1250) — matrix operand order repaired and checked by reading, not compiled, not run on hardware
//
#include <hip/hip_runtime.h>
#include <hip/hip_bf16.h>
#include <hip/hip_fp16.h>

// NAM fused kernel for MI455X (gfx1250, wave32, WMMA).
// 17 independent 1->64->32->1 MLPs, summed per sample.
// Middle layer [B,64]x[64,32] runs on v_wmma_f32_16x16x32_f16 (f32 accum).
// Layer 1 runs in packed f16 (v_pk_fma_f16 / v_pk_max_num_f16) since its
// output is consumed by an f16 WMMA anyway.

typedef __attribute__((ext_vector_type(16))) _Float16 v16h;
typedef __attribute__((ext_vector_type(8)))  float    v8f;
typedef __attribute__((ext_vector_type(4)))  float    v4f;

#define B_TOT 262144
#define F_N   17
#define H1    64
#define H2    32
#define ROWS_PER_WAVE 64           // 4 tiles of 16 rows
#define WAVES_PER_BLOCK 4
#define TPB   (WAVES_PER_BLOCK * 32)
#define ROWS_PER_BLOCK (ROWS_PER_WAVE * WAVES_PER_BLOCK)
#define XSTAGE (ROWS_PER_WAVE * F_N)   // 1088 floats per wave

#define W2_WS_HALVES (F_N * 4 * 32 * 16)   // 34816 f16 = 68 KB
#define W1_WS_HALVES (F_N * 32 * 64)       // 34816 f16 = 68 KB

// ---------------------------------------------------------------------------
// Prep 1: W2 [F,64,32] fp32 -> f16 per-lane WMMA B-fragment layout.
// Fragment (f, ks, ns): B slice K=[32ks,32ks+32), N=[16ns,16ns+16).
// 16-bit B layout: lane half h = lane>>4; column n = lane&15;
// element e -> k offset 8*h + (e&7) + 16*(e>>3).
// ---------------------------------------------------------------------------
__global__ void nam_prep_w2_f16(const float* __restrict__ W2,
                                _Float16* __restrict__ wsb) {
    int idx = blockIdx.x * blockDim.x + threadIdx.x;
    if (idx >= W2_WS_HALVES) return;
    int e    = idx & 15;
    int lane = (idx >> 4) & 31;
    int rest = idx >> 9;
    int ns   = rest & 1;
    int ks   = (rest >> 1) & 1;
    int f    = rest >> 2;
    int n    = 16 * ns + (lane & 15);
    int k    = 32 * ks + ((lane >> 4) << 3) + (e & 7) + ((e >> 3) << 4);
    wsb[idx] = (_Float16)W2[(f * H1 + k) * H2 + n];
}

// ---------------------------------------------------------------------------
// Prep 2: W1/b1 [F,64] fp32 -> f16, packed per lane in A-fragment element
// order. A 16-bit layout: lane half abase = (lane>=16 ? 8 : 0);
// slice ks element e -> k = 32*ks + abase + (e&7) + 16*(e>>3).
// Per (f,lane): 64 halves = [ w1(s=0..31) | b1(s=0..31) ], s = ks*16 + e.
// ---------------------------------------------------------------------------
__global__ void nam_prep_w1_f16(const float* __restrict__ W1,
                                const float* __restrict__ b1,
                                _Float16* __restrict__ wsw) {
    int idx = blockIdx.x * blockDim.x + threadIdx.x;
    if (idx >= F_N * 32 * 32) return;
    int s    = idx & 31;
    int lane = (idx >> 5) & 31;
    int f    = idx >> 10;
    int ks   = s >> 4;
    int e    = s & 15;
    int k    = 32 * ks + ((lane >> 4) << 3) + (e & 7) + ((e >> 3) << 4);
    size_t base = ((size_t)(f * 32 + lane)) * 64;
    wsw[base + s]      = (_Float16)W1[f * H1 + k];
    wsw[base + 32 + s] = (_Float16)b1[f * H1 + k];
}

// ---------------------------------------------------------------------------
// Main fused kernel.
// ---------------------------------------------------------------------------
__global__ void __launch_bounds__(TPB)
nam_fused_wmma(const float* __restrict__ x,
               const float* __restrict__ b2,
               const float* __restrict__ W3, const float* __restrict__ b3,
               const float* __restrict__ biasp,
               const _Float16* __restrict__ wsb,     // W2 fragments
               const _Float16* __restrict__ wsw,     // W1/b1 packed
               float* __restrict__ out) {
    __shared__ float lds_x[WAVES_PER_BLOCK][XSTAGE];

    const int lane = threadIdx.x & 31;
    const int wave = threadIdx.x >> 5;
    const int row0 = blockIdx.x * ROWS_PER_BLOCK + wave * ROWS_PER_WAVE;

    // Stage this wave's 64 rows of x (64*17 floats) into LDS, coalesced.
    const float* xsrc = x + (size_t)row0 * F_N;
    float* xl = lds_x[wave];
#pragma unroll
    for (int i = 0; i < XSTAGE / 32; ++i)           // 34 coalesced b32 loads
        xl[i * 32 + lane] = xsrc[i * 32 + lane];
    __syncthreads();

    const int half = lane >> 4;        // 0: lanes 0-15, 1: lanes 16-31
    const int n15  = lane & 15;        // column / row-in-tile index

    float partial[32];                 // [tile(4)][r(8)] column-striped sums
#pragma unroll
    for (int i = 0; i < 32; ++i) partial[i] = 0.0f;
    float b3acc = 0.0f;

    const v16h zh = {};                // f16 zeros for ReLU

    for (int f = 0; f < F_N; ++f) {
        // --- resident B fragments for this feature (4 x v16h) -------------
        const _Float16* wb = wsb + (((size_t)f * 4) * 32 + lane) * 16;
        v16h bf[4];                    // index = ks*2 + ns
#pragma unroll
        for (int q = 0; q < 4; ++q)
            bf[q] = *(const v16h*)(wb + (size_t)q * 32 * 16);
        if (f + 1 < F_N) {             // global_prefetch_b8 of next feature
            __builtin_prefetch(wsb + (((size_t)(f + 1) * 4) * 32 + lane) * 16, 0, 3);
            __builtin_prefetch(wsw + ((size_t)((f + 1) * 32 + lane)) * 64, 0, 3);
        }

        // --- per-lane W1/b1 in f16, A-element order ------------------------
        const _Float16* wp = wsw + ((size_t)(f * 32 + lane)) * 64;
        const v16h w1a = *(const v16h*)(wp);        // ks = 0
        const v16h w1b = *(const v16h*)(wp + 16);   // ks = 1
        const v16h b1a = *(const v16h*)(wp + 32);
        const v16h b1b = *(const v16h*)(wp + 48);

        // --- head / bias weights -------------------------------------------
        const float w3a = W3[f * H2 + n15];
        const float w3b = W3[f * H2 + 16 + n15];
        const float b2a = b2[f * H2 + n15];
        const float b2b = b2[f * H2 + 16 + n15];
        b3acc += b3[f];

        // --- 4 row tiles of 16 ---------------------------------------------
#pragma unroll
        for (int t4 = 0; t4 < 4; ++t4) {
            // A-fragment row M = lane&15 (both lane halves).
            const float xm = xl[(t4 * 16 + n15) * F_N + f];
            const _Float16 xh = (_Float16)xm;

            // Layer 1 in packed f16: h1 = relu(x*W1 + b1).
            const v16h a0 = __builtin_elementwise_max(xh * w1a + b1a, zh);
            const v16h a1 = __builtin_elementwise_max(xh * w1b + b1b, zh);

            // Accumulators preloaded with b2 (every row gets same bias).
            v8f c0, c1;
#pragma unroll
            for (int r = 0; r < 8; ++r) { c0[r] = b2a; c1[r] = b2b; }

            // h2[16x32] = A[16x64] x W2[64x32] + b2 : 4 WMMAs (2 K x 2 N).
            c0 = __builtin_amdgcn_wmma_f32_16x16x32_f16(false, a0, false, bf[0],
                                                        (short)0, c0, false, false);
            c0 = __builtin_amdgcn_wmma_f32_16x16x32_f16(false, a1, false, bf[2],
                                                        (short)0, c0, false, false);
            c1 = __builtin_amdgcn_wmma_f32_16x16x32_f16(false, a0, false, bf[1],
                                                        (short)0, c1, false, false);
            c1 = __builtin_amdgcn_wmma_f32_16x16x32_f16(false, a1, false, bf[3],
                                                        (short)0, c1, false, false);

            // Head: partial[row] += relu(h2) . W3   (column-striped per lane)
#pragma unroll
            for (int r = 0; r < 8; ++r) {
                partial[t4 * 8 + r] = fmaf(fmaxf(c0[r], 0.0f), w3a, partial[t4 * 8 + r]);
                partial[t4 * 8 + r] = fmaf(fmaxf(c1[r], 0.0f), w3b, partial[t4 * 8 + r]);
            }
        }
    }

    const float base_add = biasp[0] + b3acc;

    // Width-16 shfl_xor reduction per (tile, r); halves hold rows r / r+8.
    float* ol = lds_x[wave];   // reuse staging region (wave-private)
#pragma unroll
    for (int t4 = 0; t4 < 4; ++t4) {
#pragma unroll
        for (int r = 0; r < 8; ++r) {
            float v = partial[t4 * 8 + r];
            v += __shfl_xor(v, 1, 32);
            v += __shfl_xor(v, 2, 32);
            v += __shfl_xor(v, 4, 32);
            v += __shfl_xor(v, 8, 32);
            if (n15 == 0)
                ol[t4 * 16 + half * 8 + r] = v + base_add;
        }
    }
    // DS ops are in-order within a wave; coalesced global stores of 64 rows.
#pragma unroll
    for (int i = 0; i < 2; ++i)
        out[row0 + i * 32 + lane] = ol[i * 32 + lane];
}

// ---------------------------------------------------------------------------
extern "C" void kernel_launch(void* const* d_in, const int* in_sizes, int n_in,
                              void* d_out, int out_size, void* d_ws, size_t ws_size,
                              hipStream_t stream) {
    const float* x    = (const float*)d_in[0];
    const float* W1   = (const float*)d_in[1];
    const float* b1   = (const float*)d_in[2];
    const float* W2   = (const float*)d_in[3];
    const float* b2   = (const float*)d_in[4];
    const float* W3   = (const float*)d_in[5];
    const float* b3   = (const float*)d_in[6];
    const float* bias = (const float*)d_in[7];

    _Float16* wsb = (_Float16*)d_ws;                 // W2 fragments (68 KB)
    _Float16* wsw = wsb + W2_WS_HALVES;              // W1/b1 packed (68 KB)

    nam_prep_w2_f16<<<(W2_WS_HALVES + 127) / 128, 128, 0, stream>>>(W2, wsb);
    nam_prep_w1_f16<<<(F_N * 32 * 32 + 127) / 128, 128, 0, stream>>>(W1, b1, wsw);

    nam_fused_wmma<<<B_TOT / ROWS_PER_BLOCK, TPB, 0, stream>>>(
        x, b2, W3, b3, bias, wsb, wsw, (float*)d_out);
}
